// GateTypeExpertLayer_38654705664486
// MI455X (gfx1250) — compile-verified, matrix operand order checked by reading, not run on hardware
//
#include <hip/hip_runtime.h>
#include <hip/hip_bf16.h>
#include <math.h>

typedef __attribute__((ext_vector_type(16))) _Float16 v16h;
typedef __attribute__((ext_vector_type(8)))  float    v8f;

#define HD   128
#define H2   256
#define NEXP 8

// Branch-free exact-style gelu: 0.5*x*(1+erf(x/sqrt(2))) with A&S 7.1.26 erf
// (max abs err ~1.5e-7 << f16 storage quantization; no EXEC divergence).
__device__ __forceinline__ float gelu_fast(float v) {
    float s = v * 0.70710678118654752f;
    float a = fabsf(s);
    float t = 1.0f / (1.0f + 0.3275911f * a);
    float poly = t * (0.254829592f +
                 t * (-0.284496736f +
                 t * (1.421413741f +
                 t * (-1.453152027f +
                 t * 1.061405429f))));
    float erf_a = 1.0f - poly * __expf(-a * a);
    float erf_s = copysignf(erf_a, s);
    return 0.5f * v * (1.0f + erf_s);
}

// ---------------------------------------------------------------------------
// Prep A: convert x to f16 (vectorized, once per launch)
// ---------------------------------------------------------------------------
__global__ void x_to_f16_kernel(const float* __restrict__ x,
                                _Float16* __restrict__ xh, size_t total4) {
    size_t i = ((size_t)blockIdx.x * blockDim.x + threadIdx.x) * 4;
    if (i >= total4 * 4) return;
    float4 f = *(const float4*)(x + i);
    _Float16 h[4] = {(_Float16)f.x, (_Float16)f.y, (_Float16)f.z, (_Float16)f.w};
    *(uint2*)(xh + i) = *(uint2*)h;
}

// ---------------------------------------------------------------------------
// Prep B: swizzle W1/W2 into per-lane WMMA B-fragment order (f16).
// Fragment = 32 lanes x 16 halves (32B/lane, contiguous per lane).
// W1: per expert 16 ctiles x 4 kc = 64 frags.  W2: 8 ctiles x 8 kc = 64 frags.
// ---------------------------------------------------------------------------
__global__ void weight_swizzle_kernel(const float* __restrict__ W1,
                                      const float* __restrict__ W2,
                                      _Float16* __restrict__ W1f,
                                      _Float16* __restrict__ W2f) {
    int gid  = blockIdx.x * blockDim.x + threadIdx.x;   // 0 .. 2*8*64*32
    int lane = gid & 31;
    int frag = (gid >> 5) & 63;
    int e    = (gid >> 11) & 7;
    int isW2 = gid >> 14;
    int lcol  = lane & 15;
    int khalf = (lane < 16) ? 0 : 8;
    v16h v;
    if (!isW2) {
        int ct = frag >> 2, kc = frag & 3;
        int col   = ct * 16 + lcol;
        int kbase = kc * 32 + khalf;
        const float* src = W1 + (size_t)e * HD * H2;
        #pragma unroll
        for (int i = 0; i < 8; ++i) {
            v[i]     = (_Float16)src[(size_t)(kbase + i) * H2 + col];
            v[8 + i] = (_Float16)src[(size_t)(kbase + 16 + i) * H2 + col];
        }
        ((v16h*)W1f)[(size_t)(e * 64 + frag) * 32 + lane] = v;
    } else {
        int ct = frag >> 3, kc = frag & 7;
        int col   = ct * 16 + lcol;
        int kbase = kc * 32 + khalf;
        const float* src = W2 + (size_t)e * H2 * HD;
        #pragma unroll
        for (int i = 0; i < 8; ++i) {
            v[i]     = (_Float16)src[(size_t)(kbase + i) * HD + col];
            v[8 + i] = (_Float16)src[(size_t)(kbase + 16 + i) * HD + col];
        }
        ((v16h*)W2f)[(size_t)(e * 64 + frag) * 32 + lane] = v;
    }
}

// ---------------------------------------------------------------------------
// Kernel 1: scatter-mean numerator/denominator over destination nodes
// ---------------------------------------------------------------------------
__global__ void edge_scatter_kernel(const int* __restrict__ gate_type,
                                    const int* __restrict__ edge_index,
                                    const float* __restrict__ embed,
                                    float* __restrict__ seg_sum,
                                    float* __restrict__ cnt_buf,
                                    int eEdges) {
    int i = blockIdx.x * blockDim.x + threadIdx.x;
    if (i >= eEdges) return;
    int dst = edge_index[eEdges + i];   // row 1 of [2, E]
    int t   = gate_type[i];
    atomicAdd(&cnt_buf[dst], 1.0f);
    #pragma unroll
    for (int ex = 0; ex < NEXP; ++ex)
        atomicAdd(&seg_sum[(size_t)dst * NEXP + ex], embed[t * NEXP + ex]);
}

// ---------------------------------------------------------------------------
// Kernel 2: router + per-expert node-list compaction (one wave32 per node)
// ---------------------------------------------------------------------------
__global__ void router_kernel(const float* __restrict__ x,
                              const float* __restrict__ Wr,
                              const float* __restrict__ br,
                              const float* __restrict__ seg_sum,
                              const float* __restrict__ cnt_buf,
                              int* __restrict__ expert_count,
                              int* __restrict__ list_node,
                              float* __restrict__ list_gate,
                              int n) {
    int node = blockIdx.x * 8 + (threadIdx.x >> 5);
    if (node >= n) return;
    int lane = threadIdx.x & 31;

    const float* xr = x + (size_t)node * HD;
    float acc[NEXP];
    #pragma unroll
    for (int ex = 0; ex < NEXP; ++ex) acc[ex] = 0.0f;
    #pragma unroll
    for (int i = 0; i < 4; ++i) {
        int d = lane * 4 + i;
        float xv = xr[d];
        #pragma unroll
        for (int ex = 0; ex < NEXP; ++ex)
            acc[ex] += xv * Wr[d * NEXP + ex];
    }
    #pragma unroll
    for (int ex = 0; ex < NEXP; ++ex)
        for (int off = 16; off > 0; off >>= 1)
            acc[ex] += __shfl_xor(acc[ex], off, 32);

    if (lane == 0) {
        float c   = cnt_buf[node];
        float inv = 1.0f / fmaxf(c, 1.0f);
        float logits[NEXP];
        #pragma unroll
        for (int ex = 0; ex < NEXP; ++ex) {
            float g = (c > 0.0f) ? seg_sum[(size_t)node * NEXP + ex] * inv : 0.0f;
            logits[ex] = acc[ex] + br[ex] + g;
        }
        int i0 = 0;
        #pragma unroll
        for (int ex = 1; ex < NEXP; ++ex) if (logits[ex] > logits[i0]) i0 = ex;
        int i1 = (i0 == 0) ? 1 : 0;
        #pragma unroll
        for (int ex = 0; ex < NEXP; ++ex)
            if (ex != i0 && logits[ex] > logits[i1]) i1 = ex;
        float e1 = __expf(logits[i1] - logits[i0]);
        float g0 = 1.0f / (1.0f + e1);
        float g1 = e1 * g0;

        int p0 = atomicAdd(&expert_count[i0], 1);
        list_node[(size_t)i0 * n + p0] = node;
        list_gate[(size_t)i0 * n + p0] = g0;
        int p1 = atomicAdd(&expert_count[i1], 1);
        list_node[(size_t)i1 * n + p1] = node;
        list_gate[(size_t)i1 * n + p1] = g1;
    }
}

// ---------------------------------------------------------------------------
// Kernel 3: per-expert 2-layer MLP on compacted 16-node tiles, f16 WMMA.
// B fragments come pre-swizzled (1 x 32B load per lane per WMMA).
// Block: 256 threads = 8 waves (wave32). All WMMA branches block-uniform.
// ---------------------------------------------------------------------------
__global__ void expert_mlp_kernel(const _Float16* __restrict__ xh,
                                  const _Float16* __restrict__ W1f,
                                  const float* __restrict__ b1,
                                  const _Float16* __restrict__ W2f,
                                  const float* __restrict__ b2,
                                  const int* __restrict__ expert_count,
                                  const int* __restrict__ list_node,
                                  const float* __restrict__ list_gate,
                                  float* __restrict__ out,
                                  int n) {
    __shared__ _Float16 sA[16 * HD];   // 4 KB: x tile (f16)
    __shared__ _Float16 sH[16 * H2];   // 8 KB: hidden activations (f16)
    __shared__ int   s_node[16];
    __shared__ float s_gate[16];

    const int e    = blockIdx.y;
    const int tile = blockIdx.x;
    const int ecnt = expert_count[e];
    if (tile * 16 >= ecnt) return;     // block-uniform exit

    const int tid = threadIdx.x;
    if (tid < 16) {
        int idx = tile * 16 + tid;
        if (idx < ecnt) {
            s_node[tid] = list_node[(size_t)e * n + idx];
            s_gate[tid] = list_gate[(size_t)e * n + idx];
        } else {
            s_node[tid] = 0;
            s_gate[tid] = 0.0f;        // padded row contributes 0
        }
    }
    __syncthreads();

    // stage gathered f16 x rows: 16 rows x 16 chunks of 8 halves (16B each)
    {
        int r  = tid >> 4;
        int c8 = (tid & 15) * 8;
        uint4 pkt = *(const uint4*)(xh + (size_t)s_node[r] * HD + c8);
        *(uint4*)(&sA[r * HD + c8]) = pkt;
    }
    __syncthreads();

    const int wave  = tid >> 5;
    const int lane  = tid & 31;
    const int lrow  = lane & 15;                 // A: M = lane%16 (both halves)
    const int khalf = (lane < 16) ? 0 : 8;       // A/B K sub-offset per half-wave
    const int rbase = (lane < 16) ? 0 : 8;       // C/D: M base per half-wave
    const v16h* W1fe = (const v16h*)W1f + (size_t)e * 64 * 32 + lane;
    const v16h* W2fe = (const v16h*)W2f + (size_t)e * 64 * 32 + lane;

    // ---- Layer 1: A fragments hoisted, reused for both column tiles ------
    v16h afr[4];
    #pragma unroll
    for (int kc = 0; kc < 4; ++kc) {
        const int ka = kc * 32 + khalf;
        #pragma unroll
        for (int i = 0; i < 8; ++i) {
            afr[kc][i]     = sA[lrow * HD + ka + i];
            afr[kc][8 + i] = sA[lrow * HD + ka + 16 + i];
        }
    }
    #pragma unroll
    for (int cti = 0; cti < 2; ++cti) {
        const int ct = wave + cti * 8;
        v8f acc = {};
        #pragma unroll
        for (int kc = 0; kc < 4; ++kc) {
            v16h b = W1fe[(size_t)(ct * 4 + kc) * 32];
            if (kc < 3)
                __builtin_prefetch(&W1fe[(size_t)(ct * 4 + kc + 1) * 32], 0, 0);
            acc = __builtin_amdgcn_wmma_f32_16x16x32_f16(
                false, afr[kc], false, b, (short)0, acc, false, false);
        }
        const float bias = b1[e * H2 + ct * 16 + lrow];
        #pragma unroll
        for (int r = 0; r < 8; ++r) {
            float v = gelu_fast(acc[r] + bias);
            sH[(rbase + r) * H2 + ct * 16 + lrow] = (_Float16)v;
        }
    }
    __syncthreads();

    // ---- Layer 2: 8 column tiles of 16; one per wave ---------------------
    {
        const int ct = wave;
        v8f acc = {};
        #pragma unroll
        for (int kc = 0; kc < 8; ++kc) {
            const int ka = kc * 32 + khalf;
            v16h a, b;
            #pragma unroll
            for (int i = 0; i < 8; ++i) {
                a[i]     = sH[lrow * H2 + ka + i];
                a[8 + i] = sH[lrow * H2 + ka + 16 + i];
            }
            b = W2fe[(size_t)(ct * 8 + kc) * 32];
            if (kc < 7)
                __builtin_prefetch(&W2fe[(size_t)(ct * 8 + kc + 1) * 32], 0, 0);
            acc = __builtin_amdgcn_wmma_f32_16x16x32_f16(
                false, a, false, b, (short)0, acc, false, false);
        }
        const float bias = b2[e * HD + ct * 16 + lrow];
        #pragma unroll
        for (int r = 0; r < 8; ++r) {
            const int m = rbase + r;
            float v = (acc[r] + bias) * s_gate[m];
            atomicAdd(&out[(size_t)s_node[m] * HD + ct * 16 + lrow], v);
        }
    }
}

// ---------------------------------------------------------------------------
// Kernel 4: in-place LayerNorm, one wave32 per node row of 128
// ---------------------------------------------------------------------------
__global__ void layernorm_kernel(float* __restrict__ out,
                                 const float* __restrict__ gamma,
                                 const float* __restrict__ beta,
                                 int n) {
    int node = blockIdx.x * 8 + (threadIdx.x >> 5);
    if (node >= n) return;
    int lane = threadIdx.x & 31;
    float* row = out + (size_t)node * HD;

    float v[4];
    float s = 0.0f;
    #pragma unroll
    for (int i = 0; i < 4; ++i) { v[i] = row[lane * 4 + i]; s += v[i]; }
    for (int off = 16; off > 0; off >>= 1) s += __shfl_xor(s, off, 32);
    float mu = s * (1.0f / HD);

    float var = 0.0f;
    #pragma unroll
    for (int i = 0; i < 4; ++i) { float d = v[i] - mu; var += d * d; }
    for (int off = 16; off > 0; off >>= 1) var += __shfl_xor(var, off, 32);
    var *= (1.0f / HD);
    float rstd = rsqrtf(var + 1e-5f);

    #pragma unroll
    for (int i = 0; i < 4; ++i) {
        int c = lane * 4 + i;
        row[c] = (v[i] - mu) * rstd * gamma[c] + beta[c];
    }
}

// ---------------------------------------------------------------------------
extern "C" void kernel_launch(void* const* d_in, const int* in_sizes, int n_in,
                              void* d_out, int out_size, void* d_ws, size_t ws_size,
                              hipStream_t stream) {
    const float* x     = (const float*)d_in[0];
    const int*   egt   = (const int*)d_in[1];
    const int*   eidx  = (const int*)d_in[2];
    const float* embed = (const float*)d_in[3];
    const float* Wr    = (const float*)d_in[4];
    const float* br    = (const float*)d_in[5];
    const float* W1    = (const float*)d_in[6];
    const float* b1    = (const float*)d_in[7];
    const float* W2    = (const float*)d_in[8];
    const float* b2    = (const float*)d_in[9];
    const float* gamma = (const float*)d_in[10];
    const float* beta  = (const float*)d_in[11];

    const int n  = in_sizes[0] / HD;   // 100000 nodes
    const int eE = in_sizes[1];        // 1.6M edges

    // Workspace layout (~37 MB, all offsets 32B-aligned)
    char* ws = (char*)d_ws;
    float*    seg_sum      = (float*)ws;    ws += (size_t)n * NEXP * sizeof(float);
    float*    cnt_buf      = (float*)ws;    ws += (size_t)n * sizeof(float);
    int*      expert_count = (int*)ws;      ws += 64;
    int*      list_node    = (int*)ws;      ws += (size_t)NEXP * n * sizeof(int);
    float*    list_gate    = (float*)ws;    ws += (size_t)NEXP * n * sizeof(float);
    _Float16* xh           = (_Float16*)ws; ws += (size_t)n * HD * sizeof(_Float16);
    _Float16* W1f          = (_Float16*)ws; ws += (size_t)NEXP * HD * H2 * sizeof(_Float16);
    _Float16* W2f          = (_Float16*)ws;

    hipMemsetAsync(seg_sum, 0, (size_t)n * NEXP * sizeof(float), stream);
    hipMemsetAsync(cnt_buf, 0, (size_t)n * sizeof(float), stream);
    hipMemsetAsync(expert_count, 0, 64, stream);
    hipMemsetAsync(d_out, 0, (size_t)out_size * sizeof(float), stream);

    // prep: f16 conversions / weight swizzle
    const size_t total4 = (size_t)n * HD / 4;
    x_to_f16_kernel<<<(unsigned)((total4 + 255) / 256), 256, 0, stream>>>(x, xh, total4);
    weight_swizzle_kernel<<<(2 * NEXP * 64 * 32) / 256, 256, 0, stream>>>(W1, W2, W1f, W2f);

    edge_scatter_kernel<<<(eE + 255) / 256, 256, 0, stream>>>(
        egt, eidx, embed, seg_sum, cnt_buf, eE);

    router_kernel<<<(n + 7) / 8, 256, 0, stream>>>(
        x, Wr, br, seg_sum, cnt_buf, expert_count, list_node, list_gate, n);

    dim3 grid((n + 15) / 16, NEXP);
    expert_mlp_kernel<<<grid, 256, 0, stream>>>(
        xh, W1f, b1, W2f, b2, expert_count, list_node, list_gate, (float*)d_out, n);

    layernorm_kernel<<<(n + 7) / 8, 256, 0, stream>>>(
        (float*)d_out, gamma, beta, n);
}
